// FeatureExtractorLayer_979252543697
// MI455X (gfx1250) — compile-verified, matrix operand order checked by reading, not run on hardware
//
#include <hip/hip_runtime.h>
#include <cstddef>
#include <cstdint>

typedef __attribute__((ext_vector_type(16))) _Float16     v16h;
typedef __attribute__((ext_vector_type(8)))  float        v8f;
typedef __attribute__((ext_vector_type(8)))  unsigned int v8u;

#define GEMM_WAVES 8
#define GEMM_ROWS  (GEMM_WAVES * 16)

__device__ __forceinline__ unsigned pack2h(float a, float b) {
  _Float16 ha = (_Float16)a, hb = (_Float16)b;
  unsigned short ua = __builtin_bit_cast(unsigned short, ha);
  unsigned short ub = __builtin_bit_cast(unsigned short, hb);
  return (unsigned)ua | ((unsigned)ub << 16);
}

// ---------------------------------------------------------------- utility ---
__global__ void fill_f32_kernel(float* p, float v, long n) {
  long i = (long)blockIdx.x * blockDim.x + threadIdx.x;
  if (i < n) p[i] = v;
}
__global__ void fill_u32_kernel(unsigned* p, unsigned v, long n) {
  long i = (long)blockIdx.x * blockDim.x + threadIdx.x;
  if (i < n) p[i] = v;
}

// edge_comb = [edge_learned(32) | lo | hi | def_mm | edge_lp_wo_ss(5)]  -> [NE,40]
__global__ void build_ec_kernel(const float* __restrict__ el, const float* __restrict__ lo,
                                const float* __restrict__ hi, const float* __restrict__ dm,
                                const float* __restrict__ wo, float* __restrict__ ec, int nE) {
  long idx = (long)blockIdx.x * blockDim.x + threadIdx.x;
  if (idx >= (long)nE * 40) return;
  long e = idx / 40; int c = (int)(idx % 40);
  float x;
  if (c < 32) x = el[e * 32 + c];
  else if (c == 32) x = lo[e];
  else if (c == 33) x = hi[e];
  else if (c == 34) x = dm[e];
  else x = wo[e * 5 + (c - 35)];
  ec[idx] = x;
}

__global__ void count_edges_kernel(const int* __restrict__ iv, const int* __restrict__ ic,
                                   int* cv, int* cc, int nE) {
  int e = blockIdx.x * blockDim.x + threadIdx.x;
  if (e < nE) { atomicAdd(cv + iv[e], 1); atomicAdd(cc + ic[e], 1); }
}

// -------------------------------------------------- generic WMMA GEMM -------
// Y[M,N] = concat(X0[:,0:d0], X1[:,0:d1]) @ B (+bias)(relu?), B = W or W^T.
// LDS tiles staged as packed f16 pairs (k-major dwords); one wave per 16-row
// tile, 8 waves/block => 128 rows/block.
__global__ void gemm_cat2_wmma(float* __restrict__ Y, int ldy,
                               const float* __restrict__ X0, int d0,
                               const float* __restrict__ X1, int d1, int M,
                               const float* __restrict__ W, int wK, int wN, int transB,
                               const float* __restrict__ bias, int relu) {
  extern __shared__ unsigned ldsu[];
  const int K = transB ? wN : wK;
  const int N = transB ? wK : wN;
  const int KP  = (K + 31) & ~31;
  const int NP  = (N + 15) & ~15;
  const int KPH = KP >> 1;      // packed pairs per row
  const int SA  = KPH + 1;      // padded A row pitch (bank-conflict dodge)
  unsigned* ldsBu = ldsu;                 // KPH*NP
  unsigned* ldsAu = ldsu + KPH * NP;      // GEMM_ROWS*SA
  const int tid = threadIdx.x, nth = blockDim.x;
  const int rowBase = blockIdx.x * GEMM_ROWS;

  // prefetch next row-block of X0 (global_prefetch_b8 path)
  if (tid < GEMM_ROWS && (rowBase + GEMM_ROWS + tid) < M)
    __builtin_prefetch(X0 + (long)(rowBase + GEMM_ROWS + tid) * d0, 0, 1);

  for (int idx = tid; idx < KPH * NP; idx += nth) {
    int p = idx / NP, n = idx % NP;
    int k0 = 2 * p, k1 = k0 + 1;
    float w0 = 0.0f, w1 = 0.0f;
    if (n < N) {
      if (k0 < K) w0 = transB ? W[(long)n * wN + k0] : W[(long)k0 * wN + n];
      if (k1 < K) w1 = transB ? W[(long)n * wN + k1] : W[(long)k1 * wN + n];
    }
    ldsBu[p * NP + n] = pack2h(w0, w1);
  }
  for (int idx = tid; idx < GEMM_ROWS * KPH; idx += nth) {
    int r = idx / KPH, p = idx % KPH;
    int row = rowBase + r;
    float x0 = 0.0f, x1 = 0.0f;
    if (row < M) {
      int k0 = 2 * p, k1 = k0 + 1;
      if (k0 < d0) x0 = X0[(long)row * d0 + k0];
      else if (k0 - d0 < d1) x0 = X1[(long)row * d1 + (k0 - d0)];
      if (k1 < d0) x1 = X0[(long)row * d0 + k1];
      else if (k1 - d0 < d1) x1 = X1[(long)row * d1 + (k1 - d0)];
    }
    ldsAu[r * SA + p] = pack2h(x0, x1);
  }
  __syncthreads();

  const int wave = tid >> 5, lane = tid & 31, half = lane >> 4, lm = lane & 15;
  const unsigned* Arow = ldsAu + (wave * 16 + lm) * SA;
  const int row0 = rowBase + wave * 16;

  for (int ct = 0; ct < NP / 16; ++ct) {
    v8f acc = {};
    for (int kc = 0; kc < KP; kc += 32) {
      v8u au, bu;
#pragma unroll
      for (int r = 0; r < 8; ++r) {
        int k0 = (r < 4) ? (half * 8 + 2 * r) : (16 + half * 8 + 2 * (r - 4));
        int p = (kc + k0) >> 1;
        au[r] = Arow[p];
        bu[r] = ldsBu[p * NP + ct * 16 + lm];
      }
      v16h af = __builtin_bit_cast(v16h, au);
      v16h bf = __builtin_bit_cast(v16h, bu);
      acc = __builtin_amdgcn_wmma_f32_16x16x32_f16(false, af, false, bf,
                                                   (short)0, acc, false, false);
    }
    int n = ct * 16 + lm;
    if (n < N) {
      float bv = bias ? bias[n] : 0.0f;
#pragma unroll
      for (int r = 0; r < 8; ++r) {
        int m = r + 8 * half;
        int row = row0 + m;
        if (row < M) {
          float y = acc[r] + bv;
          if (relu) y = fmaxf(y, 0.0f);
          Y[(long)row * ldy + n] = y;
        }
      }
    }
  }
}

// ----------------------------------------------------- attention edge passes
__global__ void edge_logit_max_kernel(const int* __restrict__ jsrc, const int* __restrict__ idst,
                                      const float* __restrict__ q, const float* __restrict__ t,
                                      const float* __restrict__ k, const float* __restrict__ ec,
                                      float* __restrict__ logit, unsigned* __restrict__ mbits,
                                      int nE) {
  int e = blockIdx.x * (blockDim.x >> 5) + (threadIdx.x >> 5);
  if (e >= nE) return;
  int lane = threadIdx.x & 31;
  long j = jsrc[e], i = idst[e];
  float p = q[i * 64 + lane] * k[j * 64 + lane] + q[i * 64 + 32 + lane] * k[j * 64 + 32 + lane];
  p += t[i * 40 + lane] * ec[(long)e * 40 + lane];
  if (lane < 8) p += t[i * 40 + 32 + lane] * ec[(long)e * 40 + 32 + lane];
  for (int off = 16; off; off >>= 1) p += __shfl_down(p, off, 32);
  if (lane == 0) {
    float lg = p * 0.125f;
    logit[e] = lg;
    unsigned u = __float_as_uint(lg);
    u = (u & 0x80000000u) ? ~u : (u | 0x80000000u);
    atomicMax(mbits + i, u);
  }
}

__global__ void edge_accum_kernel(const int* __restrict__ jsrc, const int* __restrict__ idst,
                                  const float* __restrict__ v, const float* __restrict__ ec,
                                  const float* __restrict__ logit, const unsigned* __restrict__ mbits,
                                  float* __restrict__ s, float* __restrict__ accv,
                                  float* __restrict__ accx, int nE) {
  int e = blockIdx.x * (blockDim.x >> 5) + (threadIdx.x >> 5);
  if (e >= nE) return;
  int lane = threadIdx.x & 31;
  long j = jsrc[e], i = idst[e];
  unsigned u = mbits[i];
  unsigned bits = (u & 0x80000000u) ? (u & 0x7FFFFFFFu) : ~u;
  float m = __uint_as_float(bits);
  float a = __expf(logit[e] - m);
  if (lane == 0) atomicAdd(s + i, a);
  atomicAdd(accv + i * 64 + lane,      a * v[j * 64 + lane]);
  atomicAdd(accv + i * 64 + 32 + lane, a * v[j * 64 + 32 + lane]);
  atomicAdd(accx + i * 40 + lane,      a * ec[(long)e * 40 + lane]);
  if (lane < 8) atomicAdd(accx + i * 40 + 32 + lane, a * ec[(long)e * 40 + 32 + lane]);
}

__global__ void combine_ln_kernel(const float* __restrict__ accv, const float* __restrict__ P1,
                                  const float* __restrict__ P2, const float* __restrict__ s,
                                  const int* __restrict__ cnt, const float* __restrict__ g,
                                  const float* __restrict__ beta, float* __restrict__ out, int n) {
  int row = blockIdx.x * (blockDim.x >> 5) + (threadIdx.x >> 5);
  if (row >= n) return;
  int lane = threadIdx.x & 31;
  long b = (long)row * 64;
  float inv = 1.0f / ((s[row] + 1e-16f) * fmaxf((float)cnt[row], 1.0f));
  float y0 = (accv[b + lane] + P1[b + lane]) * inv + P2[b + lane];
  float y1 = (accv[b + 32 + lane] + P1[b + 32 + lane]) * inv + P2[b + 32 + lane];
  float tsum = y0 + y1;
  for (int off = 16; off; off >>= 1) tsum += __shfl_xor(tsum, off, 32);
  float mu = tsum * (1.0f / 64.0f);
  float dz0 = y0 - mu, dz1 = y1 - mu;
  float vv = dz0 * dz0 + dz1 * dz1;
  for (int off = 16; off; off >>= 1) vv += __shfl_xor(vv, off, 32);
  float rstd = rsqrtf(vv * (1.0f / 64.0f) + 1e-5f);
  out[b + lane]      = fmaxf(dz0 * rstd * g[lane] + beta[lane], 0.0f);
  out[b + 32 + lane] = fmaxf(dz1 * rstd * g[lane + 32] + beta[lane + 32], 0.0f);
}

// -------------------------------------------------- fused edge MLP (WMMA) ---
// per wave: 16 edges. eu_in[16,104->128]->relu(@W1+b1)[16,32]->(@W2+b2)->LN->relu
// All matrix tiles staged as packed f16 dwords.
#define EM_KPH 64          // 128/2 packed pairs, layer 1
#define EM_SA  (EM_KPH+1)  // padded A pitch
#define EM_SH  17          // padded H pitch (32/2 + 1)
__global__ void edge_mlp_kernel(const float* __restrict__ ec, const float* __restrict__ vc,
                                const float* __restrict__ cc, const int* __restrict__ jv,
                                const int* __restrict__ icn,
                                const float* __restrict__ W1, const float* __restrict__ b1,
                                const float* __restrict__ W2, const float* __restrict__ b2,
                                const float* __restrict__ g, const float* __restrict__ lnb,
                                float* __restrict__ out, int nE) {
  extern __shared__ unsigned ldsu[];
  unsigned* W1u = ldsu;                    // 64*32
  unsigned* W2u = W1u + EM_KPH * 32;       // 16*32
  unsigned* A   = W2u + 16 * 32;           // 4 waves * 16*EM_SA
  unsigned* H   = A + 4 * 16 * EM_SA;      // 4 waves * 16*EM_SH
  float*    Yf  = (float*)(H + 4 * 16 * EM_SH); // 4 waves * 16*32 f32
  const int tid = threadIdx.x;

  for (int idx = tid; idx < EM_KPH * 32; idx += blockDim.x) {
    int p = idx >> 5, n = idx & 31;
    int k0 = 2 * p, k1 = k0 + 1;
    float w0 = (k0 < 104) ? W1[k0 * 32 + n] : 0.0f;
    float w1 = (k1 < 104) ? W1[k1 * 32 + n] : 0.0f;
    W1u[idx] = pack2h(w0, w1);
  }
  for (int idx = tid; idx < 16 * 32; idx += blockDim.x) {
    int p = idx >> 5, n = idx & 31;
    W2u[idx] = pack2h(W2[(2 * p) * 32 + n], W2[(2 * p + 1) * 32 + n]);
  }

  const int wave = tid >> 5, lane = tid & 31, half = lane >> 4, lm = lane & 15;
  const long ebase = (long)blockIdx.x * 64 + wave * 16;
  unsigned* Aw = A + wave * 16 * EM_SA;
  unsigned* Hw = H + wave * 16 * EM_SH;
  _Float16* Hh = (_Float16*)Hw;
  float*    Yw = Yf + wave * 16 * 32;

  for (int idx = lane; idx < 16 * EM_KPH; idx += 32) {
    int r = idx >> 6, p = idx & 63;
    long e = ebase + r;
    float x0 = 0.0f, x1 = 0.0f;
    if (e < nE) {
      int c0 = 2 * p, c1 = c0 + 1;
      if (c0 < 40) x0 = ec[e * 40 + c0];
      else if (c0 < 72) x0 = vc[(long)jv[e] * 32 + (c0 - 40)];
      else if (c0 < 104) x0 = cc[(long)icn[e] * 32 + (c0 - 72)];
      if (c1 < 40) x1 = ec[e * 40 + c1];
      else if (c1 < 72) x1 = vc[(long)jv[e] * 32 + (c1 - 40)];
      else if (c1 < 104) x1 = cc[(long)icn[e] * 32 + (c1 - 72)];
    }
    Aw[r * EM_SA + p] = pack2h(x0, x1);
  }
  __syncthreads();

  // layer 1: [16,128] @ [128,32] -> relu -> Hh (packed f16)
#pragma unroll
  for (int ct = 0; ct < 2; ++ct) {
    v8f acc = {};
#pragma unroll
    for (int kc = 0; kc < 128; kc += 32) {
      v8u au, bu;
#pragma unroll
      for (int r = 0; r < 8; ++r) {
        int k0 = (r < 4) ? (half * 8 + 2 * r) : (16 + half * 8 + 2 * (r - 4));
        int p = (kc + k0) >> 1;
        au[r] = Aw[lm * EM_SA + p];
        bu[r] = W1u[p * 32 + ct * 16 + lm];
      }
      v16h af = __builtin_bit_cast(v16h, au);
      v16h bf = __builtin_bit_cast(v16h, bu);
      acc = __builtin_amdgcn_wmma_f32_16x16x32_f16(false, af, false, bf,
                                                   (short)0, acc, false, false);
    }
    int n = ct * 16 + lm;
    float bv = b1[n];
#pragma unroll
    for (int r = 0; r < 8; ++r) {
      int m = r + 8 * half;
      Hh[m * (2 * EM_SH) + n] = (_Float16)fmaxf(acc[r] + bv, 0.0f);
    }
  }
  __syncthreads();

  // layer 2: [16,32] @ [32,32] (K=32, one WMMA per col-tile)
  v8f a2[2];
#pragma unroll
  for (int ct = 0; ct < 2; ++ct) {
    v8u au, bu;
#pragma unroll
    for (int r = 0; r < 8; ++r) {
      int k0 = (r < 4) ? (half * 8 + 2 * r) : (16 + half * 8 + 2 * (r - 4));
      int p = k0 >> 1;
      au[r] = Hw[lm * EM_SH + p];
      bu[r] = W2u[p * 32 + ct * 16 + lm];
    }
    v16h af = __builtin_bit_cast(v16h, au);
    v16h bf = __builtin_bit_cast(v16h, bu);
    v8f acc = {};
    acc = __builtin_amdgcn_wmma_f32_16x16x32_f16(false, af, false, bf,
                                                 (short)0, acc, false, false);
    a2[ct] = acc;
  }
  __syncthreads();
#pragma unroll
  for (int ct = 0; ct < 2; ++ct) {
    int n = ct * 16 + lm;
    float bv = b2[n];
#pragma unroll
    for (int r = 0; r < 8; ++r) {
      int m = r + 8 * half;
      Yw[m * 32 + n] = a2[ct][r] + bv;
    }
  }
  __syncthreads();
  if (lane < 16) {
    long e = ebase + lane;
    if (e < nE) {
      const float* hr = Yw + lane * 32;
      float mu = 0.0f;
#pragma unroll
      for (int c2 = 0; c2 < 32; ++c2) mu += hr[c2];
      mu *= (1.0f / 32.0f);
      float var = 0.0f;
#pragma unroll
      for (int c2 = 0; c2 < 32; ++c2) { float d = hr[c2] - mu; var += d * d; }
      float rstd = rsqrtf(var * (1.0f / 32.0f) + 1e-5f);
#pragma unroll
      for (int c2 = 0; c2 < 32; ++c2)
        out[e * 32 + c2] = fmaxf((hr[c2] - mu) * rstd * g[c2] + lnb[c2], 0.0f);
    }
  }
}

// ------------------------------------------------------------------- host ---
static inline unsigned divup(long a, long b) { return (unsigned)((a + b - 1) / b); }

static void launch_gemm(hipStream_t s, float* Y, int ldy, const float* X0, int d0,
                        const float* X1, int d1, long M, const float* W, int wK, int wN,
                        int transB, const float* bias, int relu) {
  int K = transB ? wN : wK, N = transB ? wK : wN;
  int KP = (K + 31) & ~31, NP = (N + 15) & ~15;
  int KPH = KP >> 1, SA = KPH + 1;
  size_t lds = (size_t)(KPH * NP + GEMM_ROWS * SA) * sizeof(unsigned);
  gemm_cat2_wmma<<<divup(M, GEMM_ROWS), 32 * GEMM_WAVES, lds, s>>>(
      Y, ldy, X0, d0, X1, d1, (int)M, W, wK, wN, transB, bias, relu);
}

extern "C" void kernel_launch(void* const* d_in, const int* in_sizes, int n_in,
                              void* d_out, int out_size, void* d_ws, size_t ws_size,
                              hipStream_t stream) {
  const float* var_learned = (const float*)d_in[0];
  const float* var_lp      = (const float*)d_in[1];
  const float* con_learned = (const float*)d_in[2];
  const float* con_lp      = (const float*)d_in[3];
  const float* e_learned   = (const float*)d_in[4];
  const float* lo          = (const float*)d_in[5];
  const float* hi          = (const float*)d_in[6];
  const float* dm          = (const float*)d_in[7];
  const float* e_wo        = (const float*)d_in[8];
  const int*   eidx        = (const int*)d_in[9];
  // params: JAX pytree (alphabetical) flattening
  const float* cc_eW   = (const float*)d_in[10];  // con_conv.e.W  [40,64]
  const float* cc_kW   = (const float*)d_in[11];  // con_conv.k.W  [75,64]
  const float* cc_kb   = (const float*)d_in[12];
  const float* cc_qW   = (const float*)d_in[13];  // [69,64]
  const float* cc_qb   = (const float*)d_in[14];
  const float* cc_sW   = (const float*)d_in[15];  // skip [69,64]
  const float* cc_sb   = (const float*)d_in[16];
  const float* cc_vW   = (const float*)d_in[17];  // [75,64]
  const float* cc_vb   = (const float*)d_in[18];
  const float* eu_cc1W = (const float*)d_in[19];  // [69,32]
  const float* eu_cc1b = (const float*)d_in[20];
  const float* eu_cc2W = (const float*)d_in[21];  // [32,32]
  const float* eu_cc2b = (const float*)d_in[22];
  const float* eu_em1W = (const float*)d_in[23];  // [104,32]
  const float* eu_em1b = (const float*)d_in[24];
  const float* eu_em2W = (const float*)d_in[25];  // [32,32]
  const float* eu_em2b = (const float*)d_in[26];
  const float* eu_vc1W = (const float*)d_in[27];  // [75,32]
  const float* eu_vc1b = (const float*)d_in[28];
  const float* eu_vc2W = (const float*)d_in[29];  // [32,32]
  const float* eu_vc2b = (const float*)d_in[30];
  const float* lncon_b = (const float*)d_in[31];
  const float* lncon_g = (const float*)d_in[32];
  const float* lnedg_b = (const float*)d_in[33];
  const float* lnedg_g = (const float*)d_in[34];
  const float* lnvar_b = (const float*)d_in[35];
  const float* lnvar_g = (const float*)d_in[36];
  const float* vc_eW   = (const float*)d_in[37];  // var_conv.e.W  [40,64]
  const float* vc_kW   = (const float*)d_in[38];  // [69,64]
  const float* vc_kb   = (const float*)d_in[39];
  const float* vc_qW   = (const float*)d_in[40];  // [75,64]
  const float* vc_qb   = (const float*)d_in[41];
  const float* vc_sW   = (const float*)d_in[42];  // [75,64]
  const float* vc_sb   = (const float*)d_in[43];
  const float* vc_vW   = (const float*)d_in[44];  // [69,64]
  const float* vc_vb   = (const float*)d_in[45];

  const long NV = in_sizes[1] / 11;
  const long NC = in_sizes[3] / 5;
  const int  NE = in_sizes[5];
  const long NMAX = (NV > NC) ? NV : NC;

  const int* idx_var = eidx;
  const int* idx_con = eidx + NE;

  float* out_var  = (float*)d_out;                  // [NV,64]
  float* out_con  = out_var + (size_t)NV * 64;      // [NC,64]
  float* out_edge = out_con + (size_t)NC * 64;      // [NE,32]

  char* wp = (char*)d_ws;
  auto alloc = [&](size_t bytes) -> float* {
    float* p = (float*)wp;
    wp += (bytes + 255) & ~((size_t)255);
    return p;
  };
  float*    ec    = alloc((size_t)NE * 40 * 4);
  float*    logit = alloc((size_t)NE * 4);
  unsigned* mbits = (unsigned*)alloc((size_t)NMAX * 4);
  float*    sseg  = alloc((size_t)NMAX * 4);
  int*      cnt_v = (int*)alloc((size_t)NV * 4);
  int*      cnt_c = (int*)alloc((size_t)NC * 4);
  float*    accv  = alloc((size_t)NMAX * 64 * 4);
  float*    accx  = alloc((size_t)NMAX * 40 * 4);
  float*    qb    = alloc((size_t)NMAX * 64 * 4);
  float*    tb    = alloc((size_t)NMAX * 40 * 4);
  float*    kb    = alloc((size_t)NMAX * 64 * 4);
  float*    vb    = alloc((size_t)NMAX * 64 * 4);
  float*    P1    = alloc((size_t)NMAX * 64 * 4);
  float*    P2    = alloc((size_t)NMAX * 64 * 4);
  float*    vcb   = alloc((size_t)NV * 32 * 4);
  float*    ccb   = alloc((size_t)NC * 32 * 4);
  float*    tmp32 = alloc((size_t)NMAX * 32 * 4);

  build_ec_kernel<<<divup((long)NE * 40, 256), 256, 0, stream>>>(e_learned, lo, hi, dm, e_wo, ec, NE);
  fill_u32_kernel<<<divup(NV, 256), 256, 0, stream>>>((unsigned*)cnt_v, 0u, NV);
  fill_u32_kernel<<<divup(NC, 256), 256, 0, stream>>>((unsigned*)cnt_c, 0u, NC);
  count_edges_kernel<<<divup(NE, 256), 256, 0, stream>>>(idx_var, idx_con, cnt_v, cnt_c, NE);

  auto run_conv = [&](long nDst, long nSrc,
                      const float* Xd0, int dd0, const float* Xd1, int dd1,
                      const float* Xs0, int ds0, const float* Xs1, int ds1,
                      const int* jsrc, const int* idst, const int* cntDst,
                      const float* We, const float* Wk, const float* bk,
                      const float* Wq, const float* bq, const float* Wsk, const float* bsk,
                      const float* Wv, const float* bvv,
                      const float* lng, const float* lnbb, float* outDst) {
    launch_gemm(stream, qb, 64, Xd0, dd0, Xd1, dd1, nDst, Wq, dd0 + dd1, 64, 0, bq, 0);
    launch_gemm(stream, tb, 40, qb, 64, nullptr, 0, nDst, We, 40, 64, 1, nullptr, 0);  // t=q@We^T
    launch_gemm(stream, kb, 64, Xs0, ds0, Xs1, ds1, nSrc, Wk, ds0 + ds1, 64, 0, bk, 0);
    launch_gemm(stream, vb, 64, Xs0, ds0, Xs1, ds1, nSrc, Wv, ds0 + ds1, 64, 0, bvv, 0);
    fill_u32_kernel<<<divup(nDst, 256), 256, 0, stream>>>(mbits, 0u, nDst);
    fill_f32_kernel<<<divup(nDst, 256), 256, 0, stream>>>(sseg, 0.0f, nDst);
    fill_f32_kernel<<<divup(nDst * 64, 256), 256, 0, stream>>>(accv, 0.0f, nDst * 64);
    fill_f32_kernel<<<divup(nDst * 40, 256), 256, 0, stream>>>(accx, 0.0f, nDst * 40);
    edge_logit_max_kernel<<<divup(NE, 8), 256, 0, stream>>>(jsrc, idst, qb, tb, kb, ec, logit, mbits, NE);
    edge_accum_kernel<<<divup(NE, 8), 256, 0, stream>>>(jsrc, idst, vb, ec, logit, mbits, sseg, accv, accx, NE);
    launch_gemm(stream, P1, 64, accx, 40, nullptr, 0, nDst, We, 40, 64, 0, nullptr, 0);  // accx@We
    launch_gemm(stream, P2, 64, Xd0, dd0, Xd1, dd1, nDst, Wsk, dd0 + dd1, 64, 0, bsk, 0);
    combine_ln_kernel<<<divup(nDst, 8), 256, 0, stream>>>(accv, P1, P2, sseg, cntDst, lng, lnbb, outDst, (int)nDst);
  };

  // con update: src=var_comb[75], dst=con_comb[69]
  run_conv(NC, NV, con_learned, 64, con_lp, 5, var_learned, 64, var_lp, 11,
           idx_var, idx_con, cnt_c,
           cc_eW, cc_kW, cc_kb, cc_qW, cc_qb, cc_sW, cc_sb, cc_vW, cc_vb,
           lncon_g, lncon_b, out_con);

  // var update: src=con_comb2[69] (con_new|con_lp), dst=var_comb[75]
  run_conv(NV, NC, var_learned, 64, var_lp, 11, out_con, 64, con_lp, 5,
           idx_con, idx_var, cnt_v,
           vc_eW, vc_kW, vc_kb, vc_qW, vc_qb, vc_sW, vc_sb, vc_vW, vc_vb,
           lnvar_g, lnvar_b, out_var);

  // edge update MLPs
  launch_gemm(stream, tmp32, 32, out_var, 64, var_lp, 11, NV, eu_vc1W, 75, 32, 0, eu_vc1b, 1);
  launch_gemm(stream, vcb, 32, tmp32, 32, nullptr, 0, NV, eu_vc2W, 32, 32, 0, eu_vc2b, 1);
  launch_gemm(stream, tmp32, 32, out_con, 64, con_lp, 5, NC, eu_cc1W, 69, 32, 0, eu_cc1b, 1);
  launch_gemm(stream, ccb, 32, tmp32, 32, nullptr, 0, NC, eu_cc2W, 32, 32, 0, eu_cc2b, 1);

  size_t emlp_lds = (size_t)(EM_KPH * 32 + 16 * 32 + 4 * 16 * EM_SA + 4 * 16 * EM_SH) * 4
                  + (size_t)(4 * 16 * 32) * 4;
  edge_mlp_kernel<<<divup(NE, 64), 128, emlp_lds, stream>>>(
      ec, vcb, ccb, idx_var, idx_con, eu_em1W, eu_em1b, eu_em2W, eu_em2b,
      lnedg_g, lnedg_b, out_edge, NE);
}
